// HyenaFilter2D_2911987826730
// MI455X (gfx1250) — matrix operand order repaired and checked by reading, not compiled, not run on hardware
//
#include <hip/hip_runtime.h>
#include <hip/hip_bf16.h>

typedef __attribute__((ext_vector_type(16))) __bf16 v16bf;
typedef __attribute__((ext_vector_type(8)))  __bf16 v8bf;
typedef __attribute__((ext_vector_type(8)))  float  v8f;

#define D_MODEL 64
#define ORDER   16
#define LL      256
#define KW      64
#define SP      264                   // padded LDS row stride (elements): 528 B, 16B-aligned
#define TAB_ELEMS (ORDER*256*256)     // bf16 elements per Toeplitz table (2 MB)

// round-to-nearest-even f32 -> bf16 (prep kernel only)
__device__ __forceinline__ unsigned short f2bf(float f){
  unsigned u = __float_as_uint(f);
  u += 0x7FFFu + ((u >> 16) & 1u);
  return (unsigned short)(u >> 16);
}

__device__ __forceinline__ v8f wmma_bf16(v16bf a, v16bf b, v8f c){
  // D = A(16x32) * B(32x16) + C, f32 accumulate
  return __builtin_amdgcn_wmma_f32_16x16x32_bf16(false, a, false, b, (short)0, c, false, false);
}

__device__ __forceinline__ v16bf mk16(v8bf lo, v8bf hi){
  union { v16bf v; v8bf h[2]; } u; u.h[0]=lo; u.h[1]=hi; return u.v;
}

// ---------------------------------------------------------------------------
// Kernel 1: implicit-MLP filter + materialize Toeplitz tables (shared, 4 MB)
//   ws[0   .. 2MB) : R[o][j][j'] = b_o[j-j']   (banded, width 64)
//   ws[2MB .. 4MB) : T[o][i][i'] = hh[i-i'][o] (lower triangular)
// ---------------------------------------------------------------------------
__global__ __launch_bounds__(256) void hyena_filter_prep(
    const float* __restrict__ zw, const float* __restrict__ zh,
    const float* __restrict__ W0, const float* __restrict__ b0,
    const float* __restrict__ freq,
    const float* __restrict__ W1, const float* __restrict__ b1,
    const float* __restrict__ W2, const float* __restrict__ b2,
    const float* __restrict__ Wf, const float* __restrict__ deltas,
    const int* __restrict__ xp, const int* __restrict__ yp,
    unsigned short* __restrict__ ws)
{
  __shared__ float a_sh[256][ORDER];
  __shared__ float scal[D_MODEL];
  const int p = threadIdx.x;          // position index 0..255

  if (p < D_MODEL) {
    float x = (float)xp[0], y = (float)yp[0];
    float dx = x - (float)D_MODEL * 0.5f;
    float dy = y - (float)LL * 0.5f;
    float num = dx*dx + dy*dy;
    float d = fabsf(deltas[p]);
    scal[p] = expf(-num / (2.0f * d * d));
  }

  // ---- 3-layer sin MLP -> hh[p][0..15] ----
  float feats[10];
  #pragma unroll
  for (int d = 0; d < 5; ++d){ feats[d] = zw[p*5+d]; feats[5+d] = zh[p*5+d]; }

  float h[ORDER], hn[ORDER];
  #pragma unroll
  for (int o = 0; o < ORDER; ++o){
    float s = b0[o];
    #pragma unroll
    for (int d = 0; d < 10; ++d) s += feats[d] * W0[o*10+d];
    h[o] = sinf(freq[o] * s);
  }
  #pragma unroll
  for (int o = 0; o < ORDER; ++o){
    float s = b1[o];
    #pragma unroll
    for (int m = 0; m < ORDER; ++m) s += h[m] * W1[o*ORDER+m];
    hn[o] = sinf(freq[o] * s);
  }
  #pragma unroll
  for (int o = 0; o < ORDER; ++o){
    float s = b2[o];
    #pragma unroll
    for (int m = 0; m < ORDER; ++m) s += hn[m] * W2[o*ORDER+m];
    h[o] = sinf(freq[o] * s);
  }
  #pragma unroll
  for (int o = 0; o < ORDER; ++o) a_sh[p][o] = h[o];
  __syncthreads();

  unsigned short* Rg = ws;
  unsigned short* Tg = ws + TAB_ELEMS;

  // R[o][j = p][j']: nonzero only for q = j - j' in [0,64)
  for (int o = 0; o < ORDER; ++o){
    unsigned short* row = Rg + (o*256 + p)*256;
    for (int j = 0; j < 256; j += 2){
      int q0 = p - j, q1 = p - j - 1;
      unsigned lo = (q0 >= 0 && q0 < KW) ? (unsigned)f2bf(Wf[q0*ORDER+o]*scal[q0]) : 0u;
      unsigned hi = (q1 >= 0 && q1 < KW) ? (unsigned)f2bf(Wf[q1*ORDER+o]*scal[q1]) : 0u;
      *(unsigned*)(row + j) = lo | (hi << 16);
    }
  }
  // T[o][i = p][i']: nonzero only for i' <= i
  for (int o = 0; o < ORDER; ++o){
    unsigned short* row = Tg + (o*256 + p)*256;
    for (int ip = 0; ip < 256; ip += 2){
      unsigned lo = (ip     <= p) ? (unsigned)f2bf(a_sh[p-ip  ][o]) : 0u;
      unsigned hi = (ip + 1 <= p) ? (unsigned)f2bf(a_sh[p-ip-1][o]) : 0u;
      *(unsigned*)(row + ip) = lo | (hi << 16);
    }
  }
}

// ---------------------------------------------------------------------------
// Kernel 2: per-image  Y = sum_o T_o * (U * R_o)  + U*bias
// 256 blocks (one per b,c image), 512 threads = 16 waves. Wave w owns
// output row-tile w (16 rows) with 16 persistent f32 accumulator tiles.
// All K loops compile-time bounded; per-tile operands batch-loaded before
// the WMMA chain so load latency overlaps matrix ops.
// ---------------------------------------------------------------------------
__global__ __launch_bounds__(512, 1) void hyena_conv(
    const float* __restrict__ in, const float* __restrict__ bias,
    const unsigned short* __restrict__ ws, float* __restrict__ out)
{
  extern __shared__ char smem[];
  __bf16* uS = (__bf16*)smem;       // image, bf16  [256][SP]
  __bf16* wS = uS + 256*SP;         // W_o = U*R_o  [256][SP]

  const int tid  = threadIdx.x;
  const int lane = tid & 31;
  const int w    = tid >> 5;        // wave id 0..15  == output row tile
  const int lm   = lane & 15;       // M (or K-row) index inside a tile
  const int lh   = lane >> 4;       // which K-half this lane holds (A layout)

  const int img  = blockIdx.x;                      // b*64 + c
  const float bv = bias[img & (D_MODEL-1)];
  const float* src = in  + (size_t)img * (256*256);
  float*       dst = out + (size_t)img * (256*256);

  const __bf16* Rg = (const __bf16*)ws;
  const __bf16* Tg = Rg + TAB_ELEMS;

  // ---- Phase A: stage image into LDS as bf16 (vector loads, packed stores)
  {
    const float4* s4 = (const float4*)src;
    for (int t = tid; t < 256*64; t += 512){
      float4 v = s4[t];
      int row = t >> 6, c4 = (t & 63) << 2;
      union { uint2 u; __bf16 h[4]; } pk;
      pk.h[0] = (__bf16)v.x; pk.h[1] = (__bf16)v.y;
      pk.h[2] = (__bf16)v.z; pk.h[3] = (__bf16)v.w;
      *(uint2*)(uS + row*SP + c4) = pk.u;
    }
  }
  __syncthreads();

  v8f acc[16];
  #pragma unroll
  for (int jt = 0; jt < 16; ++jt)
    #pragma unroll
    for (int r = 0; r < 8; ++r) acc[jt][r] = 0.f;

  const int khi2 = w >> 1;          // stage-2 triangular K bound (wave-uniform)

  for (int o = 0; o < ORDER; ++o){
    const __bf16* Ro = Rg + o*(256*256);
    const __bf16* To = Tg + o*(256*256) + (16*w + lm)*256;

    // Warm WGP-level cache for the next o's table slices (high locality).
    if (o + 1 < ORDER){
      __builtin_prefetch(Ro + 256*256 + lane*256, 0, 3);
      __builtin_prefetch(To + 256*256, 0, 3);
    }

    // ---- Stage 1: W_o = U x R_o.  Band of width 64 => K chunks are exactly
    // {jt/2-2, jt/2-1, jt/2} clamped at 0; all bounds compile-time.
    // Batch all fragment loads, then run the chained WMMAs.
    #pragma unroll
    for (int jt = 0; jt < 16; ++jt){
      const int j0 = 16*jt;
      const int kb = jt >> 1;                 // top K-chunk of the band
      v16bf aF[3], bF[3];
      #pragma unroll
      for (int d = 0; d < 3; ++d){
        const int kc = kb - 2 + d;            // constant after unroll
        if (kc >= 0){
          const __bf16* ua = uS + (16*w + lm)*SP + 32*kc + 8*lh;
          aF[d] = mk16(*(const v8bf*)ua, *(const v8bf*)(ua + 16));
          bF[d] = mk16(*(const v8bf*)(Ro + (j0 + lm)*256 + 32*kc + 8*lh), *(const v8bf*)(Ro + (j0 + lm)*256 + 32*kc + 8*lh + 16));
        }
      }
      v8f c;
      #pragma unroll
      for (int r = 0; r < 8; ++r) c[r] = 0.f;
      #pragma unroll
      for (int d = 0; d < 3; ++d){
        const int kc = kb - 2 + d;
        if (kc >= 0) c = wmma_bf16(aF[d], bF[d], c);
      }
      #pragma unroll
      for (int r = 0; r < 8; ++r)
        wS[(j0 + lm)*SP + 16*w + r + 8*lh] = (__bf16)c[r];
    }
    __syncthreads();

    // ---- Stage 2: Y += T_o x W_o  (lower triangular: kc <= w/2).
    // 8 static blocks (wave-uniform guard): one A fragment feeding 16
    // straight-line WMMAs with independent accumulators.
    #pragma unroll
    for (int kc = 0; kc < 8; ++kc){
      if (kc <= khi2){
        const __bf16* ta = To + 32*kc + 8*lh;
        v16bf a = mk16(*(const v8bf*)ta, *(const v8bf*)(ta + 16));
        #pragma unroll
        for (int jt = 0; jt < 16; ++jt){
          const __bf16* wb = wS + (16*jt + lm)*SP + 32*kc + 8*lh;
          v16bf b = mk16(*(const v8bf*)wb, *(const v8bf*)(wb + 16));
          acc[jt] = wmma_bf16(a, b, acc[jt]);
        }
      }
    }
    __syncthreads();
  }

  // ---- Epilogue: out = Y + in*bias  (C/D layout: m = r + 8*lh, n = lm)
  #pragma unroll
  for (int jt = 0; jt < 16; ++jt){
    #pragma unroll
    for (int r = 0; r < 8; ++r){
      int i = 16*w + r + 8*lh;
      int j = 16*jt + lm;
      size_t idx = (size_t)i*256 + j;
      dst[idx] = acc[jt][r] + src[idx]*bv;
    }
  }
}

// ---------------------------------------------------------------------------
extern "C" void kernel_launch(void* const* d_in, const int* in_sizes, int n_in,
                              void* d_out, int out_size, void* d_ws, size_t ws_size,
                              hipStream_t stream)
{
  (void)in_sizes; (void)n_in; (void)out_size; (void)ws_size;

  const float* input  = (const float*)d_in[0];
  const float* bias   = (const float*)d_in[1];
  const float* zw     = (const float*)d_in[2];
  const float* zh     = (const float*)d_in[3];
  const float* W0     = (const float*)d_in[4];
  const float* b0     = (const float*)d_in[5];
  const float* freq   = (const float*)d_in[6];
  const float* W1     = (const float*)d_in[7];
  const float* b1     = (const float*)d_in[8];
  const float* W2     = (const float*)d_in[9];
  const float* b2     = (const float*)d_in[10];
  const float* Wf     = (const float*)d_in[11];
  const float* deltas = (const float*)d_in[12];
  const int*   xp     = (const int*)d_in[13];
  const int*   yp     = (const int*)d_in[14];

  unsigned short* tab = (unsigned short*)d_ws;   // 4 MB: R (2MB) + T (2MB)

  hyena_filter_prep<<<1, 256, 0, stream>>>(zw, zh, W0, b0, freq,
                                           W1, b1, W2, b2, Wf, deltas,
                                           xp, yp, tab);

  const size_t smem = 2u * 256u * SP * sizeof(unsigned short);  // 270,336 B < 320 KB
  hyena_conv<<<256, 512, smem, stream>>>(input, bias, tab, (float*)d_out);
}
